// SequentialROUNDModel_13469017440916
// MI455X (gfx1250) — matrix operand order repaired and checked by reading, not compiled, not run on hardware
//
#include <hip/hip_runtime.h>
#include <hip/hip_bf16.h>
#include <math.h>

typedef __attribute__((ext_vector_type(16))) _Float16 v16h;
typedef __attribute__((ext_vector_type(8)))  float    v8f;
typedef __attribute__((ext_vector_type(4)))  float    v4f;
typedef __attribute__((ext_vector_type(4)))  unsigned int v4u;
typedef __attribute__((ext_vector_type(8)))  int      v8i;
typedef __attribute__((ext_vector_type(4)))  int      v4i;

#define B_   256
#define S_   512
#define D_   32
#define H_   512
#define K6   3072
#define C_   10
#define NT   32            // N tiles = H/16
#define KT   96            // K tiles = 3072/32
#define WGROWS 16          // rows of B per workgroup (one M tile)
#define NWG  (B_ / WGROWS) // 16 workgroups
#define THREADS 256        // 8 waves (wave32)

// LDS layout (bytes):
//   ph      16x512 f32            32768
//   feats   16x3072 f16 A-frags   98304
//   be,bd   2x512 f32              4096
//   WeLds   32 B-frags f16        32768  (loaded once via TDM)
//   wdStage 8 waves x 2 x 4KB     65536  (TDM double buffers)
#define PH_F32   (WGROWS * H_)   // 8192 floats
#define FEATS_H  (WGROWS * K6)   // 49152 halves
#define WE_H     (NT * 512)      // 16384 halves
#define STAGE_H  (8 * 2 * 2048)  // 32768 halves
#define SMEM_BYTES (PH_F32 * 4 + FEATS_H * 2 + H_ * 4 + H_ * 4 + WE_H * 2 + STAGE_H * 2)

// --------------------------------------------------------------------------
// TDM: 1-D tensor_load_to_lds of `bytes` (multiple of 8, 8B aligned) from
// global to LDS. Descriptor per CDNA5 ISA:
//   group0: count=1 | lds_addr[63:32] | global_addr[120:64] | type=2[127:126]
//   group1: data_size=3 (8B) | tensor_dim0 = tile_dim0 = nelems | 1-D tile
//   groups 2/3 (+extra group on this toolchain): zeros (dims 0 -> unused)
__device__ __forceinline__ void tdm_load_1d(unsigned ldsOff, const void* gptr,
                                            unsigned bytes) {
  unsigned long long ga = (unsigned long long)(uintptr_t)gptr;
  unsigned elems = bytes >> 3;   // data_size units (8B)
  v4u g0;
  g0[0] = 1u;                                            // count=1
  g0[1] = ldsOff;                                        // lds_addr
  g0[2] = (unsigned)(ga & 0xFFFFFFFFu);                  // global_addr[31:0]
  g0[3] = (unsigned)((ga >> 32) & 0x1FFFFFFu) | (2u << 30); // addr[56:32]|type=2
  v8i g1;
  g1[0] = (int)(3u << 16);                               // data_size=8B
  g1[1] = (int)((elems & 0xFFFFu) << 16);                // tensor_dim0[15:0]
  g1[2] = (int)((elems >> 16) & 0xFFFFu);                // tensor_dim0[31:16]
  g1[3] = (int)((elems & 0xFFFFu) << 16);                // tile_dim0
  g1[4] = 0;                                             // tile_dim1/2 = 0
  g1[5] = (int)elems;                                    // dim0 stride (unused)
  g1[6] = 0;
  g1[7] = 0;
  v4i z4 = {0, 0, 0, 0};
  v8i z8 = {0, 0, 0, 0, 0, 0, 0, 0};
  __builtin_amdgcn_tensor_load_to_lds(g0, g1, z4, z4, z8, 0);
}

__device__ __forceinline__ unsigned ldsOffsetOf(const void* p) {
  // flat LDS aperture keeps the LDS byte offset in addr[31:0]
  return (unsigned)(uintptr_t)p;
}

// --------------------------------------------------------------------------
// Scatter one f16 value (row M, feature column k = kb+col) into the 16-bit
// A-matrix 16x32 fragment layout (ISA 7.12.2).
__device__ __forceinline__ void scatterA(_Float16* feats, int kb, int row,
                                         int col, float v) {
  int r    = col & 31;
  int kt   = (kb + col) >> 5;
  int hi   = (r >> 3) & 1;
  int lane = row + hi * 16;
  int i    = (r & 7) + ((r >> 4) & 1) * 8;
  feats[kt * 512 + lane * 16 + i] = (_Float16)v;
}

// --------------------------------------------------------------------------
// Pack Wd [H, 6H] f32 into f16 B-fragments (B(k,n) = Wd[n,k]).
__global__ void pack_Wd_kernel(const float* __restrict__ Wd,
                               _Float16* __restrict__ out) {
  int idx = blockIdx.x * blockDim.x + threadIdx.x;
  if (idx >= KT * NT * 512) return;
  int f = idx >> 9, s = idx & 511;
  int lane = s >> 4, i = s & 15;
  int kt = f >> 5, nt = f & 31;
  int n = nt * 16 + (lane & 15);
  int k = kt * 32 + (i & 7) + ((i >> 3) << 4) + ((lane >= 16) ? 8 : 0);
  out[idx] = (_Float16)Wd[(size_t)n * K6 + k];
}

// Pack We [H, D] f32 into f16 B-fragments (single K=32 tile).
__global__ void pack_We_kernel(const float* __restrict__ We,
                               _Float16* __restrict__ out) {
  int idx = blockIdx.x * blockDim.x + threadIdx.x;
  if (idx >= NT * 512) return;
  int nt = idx >> 9, s = idx & 511;
  int lane = s >> 4, i = s & 15;
  int n = nt * 16 + (lane & 15);
  int k = (i & 7) + ((i >> 3) << 4) + ((lane >= 16) ? 8 : 0);
  out[idx] = (_Float16)We[(size_t)n * D_ + k];
}

// --------------------------------------------------------------------------
// Persistent scan kernel: one workgroup owns 16 batch rows for all 512 steps.
__global__ __launch_bounds__(THREADS, 1)
void round_scan_kernel(const float* __restrict__ x,
                       const float* __restrict__ be,
                       const float* __restrict__ bd,
                       const _Float16* __restrict__ Wdpk,
                       const _Float16* __restrict__ Wepk,
                       float* __restrict__ hist /* [S,B,H] */) {
  extern __shared__ char smemRaw[];
  float*    phLds   = (float*)smemRaw;                 // 16 x 512 f32
  _Float16* feats   = (_Float16*)(phLds + PH_F32);     // A fragments
  float*    beLds   = (float*)(feats + FEATS_H);
  float*    bdLds   = beLds + H_;
  _Float16* WeLds   = (_Float16*)(bdLds + H_);         // We B fragments
  _Float16* wdStage = WeLds + WE_H;                    // per-wave TDM buffers

  const int tid    = threadIdx.x;
  const int lane   = tid & 31;
  const int wave   = tid >> 5;
  const int l16    = lane & 15;
  const int hiHalf = lane >> 4;
  const int bBase  = blockIdx.x * WGROWS;

  // One-time: DMA the packed We fragments into LDS (wave 0), zero ph.
  if (wave == 0) {
    tdm_load_1d(ldsOffsetOf(WeLds), Wepk, WE_H * 2);
  }
  for (int e = tid; e < PH_F32; e += THREADS) phLds[e] = 0.0f;
  for (int e = tid; e < H_; e += THREADS) { beLds[e] = be[e]; bdLds[e] = bd[e]; }
  if (wave == 0) __builtin_amdgcn_s_wait_tensorcnt(0);
  __syncthreads();

  const int xrow = bBase + l16;          // A-layout M = lane&15
  const int kb0  = hiHalf ? 8 : 0;

  // Per-wave double-buffered staging of this wave's 4-fragment Wd strip.
  _Float16*      myStage    = wdStage + wave * 4096;            // 2 x 2048 halves
  const unsigned myStageOff = ldsOffsetOf(myStage);
  const char*    WdB        = (const char*)Wdpk;
  const unsigned strip      = (unsigned)wave * 4096u;           // byte off in slab

  for (int t = 0; t < S_; ++t) {
    // Prologue: DMA K-tiles 0 and 1; overlaps with trig phases below.
    tdm_load_1d(myStageOff,         WdB + (size_t)(0 * NT) * 1024 + strip, 4096);
    tdm_load_1d(myStageOff + 4096u, WdB + (size_t)(1 * NT) * 1024 + strip, 4096);

    // ---- phase 1: pt = x_t @ We^T (K=32 WMMA), scatter cos/sin(pt+be) ----
    const float* xr = x + ((size_t)xrow * S_ + t) * D_;
    v4f f0 = *(const v4f*)(xr + kb0);
    v4f f1 = *(const v4f*)(xr + kb0 + 4);
    v4f f2 = *(const v4f*)(xr + kb0 + 16);
    v4f f3 = *(const v4f*)(xr + kb0 + 20);
    v16h ax;
#pragma unroll
    for (int i = 0; i < 4; ++i) {
      ax[i]      = (_Float16)f0[i];
      ax[4 + i]  = (_Float16)f1[i];
      ax[8 + i]  = (_Float16)f2[i];
      ax[12 + i] = (_Float16)f3[i];
    }
#pragma unroll
    for (int j = 0; j < 4; ++j) {
      int  nt = wave * 4 + j;
      v16h bw = *(const v16h*)(WeLds + (size_t)nt * 512 + lane * 16);
      v8f  acc = {};
      acc = __builtin_amdgcn_wmma_f32_16x16x32_f16(false, ax, false, bw,
                                                   (short)0, acc, false, false);
      int col = nt * 16 + l16;
#pragma unroll
      for (int v = 0; v < 8; ++v) {
        int   m = v + hiHalf * 8;
        float q = acc[v] + beLds[col];
        scatterA(feats, 2048, m, col, __cosf(q));
        scatterA(feats, 2560, m, col, __sinf(q));
      }
    }

    // ---- phase 2: scatter cos/sin(ph), cos/sin(ph/2) ----
    for (int e = tid; e < WGROWS * H_; e += THREADS) {
      int   row = e >> 9;
      int   col = e & (H_ - 1);
      float p   = phLds[e];
      scatterA(feats, 0,    row, col, __cosf(p));
      scatterA(feats, 512,  row, col, __sinf(p));
      scatterA(feats, 1024, row, col, __cosf(0.5f * p));
      scatterA(feats, 1536, row, col, __sinf(0.5f * p));
    }
    __syncthreads();

    // ---- phase 3: ph_new = ph + feats @ Wd^T + bd (TDM-fed WMMA pipe) ----
    v8f acc[4];
#pragma unroll
    for (int j = 0; j < 4; ++j) {
      int   col = (wave * 4 + j) * 16 + l16;
      float bdv = bdLds[col];
#pragma unroll
      for (int v = 0; v < 8; ++v)
        acc[j][v] = phLds[(v + hiHalf * 8) * H_ + col] + bdv;
    }
    for (int kt = 0; kt < KT; ++kt) {
      v16h af = *(const v16h*)(feats + kt * 512 + lane * 16);
      // ensure DMA for K-tile kt has landed (in-order TDM completion)
      if (kt >= KT - 1) __builtin_amdgcn_s_wait_tensorcnt(0);
      else              __builtin_amdgcn_s_wait_tensorcnt(1);
      const _Float16* bp = myStage + (kt & 1) * 2048;
#pragma unroll
      for (int j = 0; j < 4; ++j) {
        v16h bf = *(const v16h*)(bp + (size_t)j * 512 + lane * 16);
        acc[j] = __builtin_amdgcn_wmma_f32_16x16x32_f16(false, af, false, bf,
                                                        (short)0, acc[j],
                                                        false, false);
      }
      // refill the buffer just consumed with K-tile kt+2
      if (kt + 2 < KT)
        tdm_load_1d(myStageOff + (unsigned)(kt & 1) * 4096u,
                    WdB + (size_t)((kt + 2) * NT) * 1024 + strip, 4096);
    }
    __syncthreads();  // all phLds/feats reads complete before overwrite

    // ---- phase 4: write back ph and Hhist[t] ----
    float* histT = hist + (size_t)t * (B_ * H_) + (size_t)bBase * H_;
#pragma unroll
    for (int j = 0; j < 4; ++j) {
      int col = (wave * 4 + j) * 16 + l16;
#pragma unroll
      for (int v = 0; v < 8; ++v) {
        int m = v + hiHalf * 8;
        phLds[m * H_ + col]         = acc[j][v];
        histT[(size_t)m * H_ + col] = acc[j][v];
      }
    }
    __syncthreads();
  }
}

// --------------------------------------------------------------------------
// Readout: logits = [cos ph, sin ph, cos .5ph, sin .5ph, ph] @ Wr^T + br.
__global__ void round_readout_kernel(const float* __restrict__ hist,
                                     const float* __restrict__ Wr,
                                     const float* __restrict__ br,
                                     float* __restrict__ logits) {
  int idx = blockIdx.x * blockDim.x + threadIdx.x;
  if (idx >= B_ * C_) return;
  int b = idx / C_, c = idx % C_;
  const float* ph = hist + (size_t)(S_ - 1) * B_ * H_ + (size_t)b * H_;
  const float* w  = Wr + (size_t)c * (5 * H_);
  float s = br[c];
  for (int h = 0; h < H_; ++h) {
    float p = ph[h];
    s += __cosf(p) * w[h] + __sinf(p) * w[H_ + h] +
         __cosf(0.5f * p) * w[2 * H_ + h] + __sinf(0.5f * p) * w[3 * H_ + h] +
         p * w[4 * H_ + h];
  }
  logits[idx] = s;
}

// --------------------------------------------------------------------------
extern "C" void kernel_launch(void* const* d_in, const int* in_sizes, int n_in,
                              void* d_out, int out_size, void* d_ws,
                              size_t ws_size, hipStream_t stream) {
  (void)in_sizes; (void)n_in; (void)out_size; (void)ws_size;
  const float* x  = (const float*)d_in[0];
  const float* We = (const float*)d_in[1];
  const float* be = (const float*)d_in[2];
  const float* Wd = (const float*)d_in[3];
  const float* bd = (const float*)d_in[4];
  const float* Wr = (const float*)d_in[5];
  const float* br = (const float*)d_in[6];

  float* logits = (float*)d_out;            // [B, C]
  float* hist   = (float*)d_out + B_ * C_;  // [S, B, H]

  _Float16* Wdpk = (_Float16*)d_ws;                      // 3 MB packed B-frags
  _Float16* Wepk = Wdpk + (size_t)KT * NT * 512;         // 32 KB

  pack_Wd_kernel<<<(KT * NT * 512 + 255) / 256, 256, 0, stream>>>(Wd, Wdpk);
  pack_We_kernel<<<(NT * 512 + 255) / 256, 256, 0, stream>>>(We, Wepk);
  round_scan_kernel<<<NWG, THREADS, SMEM_BYTES, stream>>>(x, be, bd, Wdpk,
                                                          Wepk, hist);
  round_readout_kernel<<<(B_ * C_ + 255) / 256, 256, 0, stream>>>(hist, Wr, br,
                                                                  logits);
}
